// BaseDecoder_13752485282317
// MI455X (gfx1250) — matrix operand order repaired and checked by reading, not compiled, not run on hardware
//
#include <hip/hip_runtime.h>

// ---------------- problem constants ----------------
#define B_      128
#define K_      36
#define VDIM_   2048
#define HID_    1024
#define EMBED_  512
#define NTOKEN_ 20000
#define MAXLEN_ 20
#define XDIM_   2560   // EMBED + VDIM
#define G3_     3072   // 3*HID
#define NT_     4      // N-tiles (16 cols each) per wave in the WMMA GEMM

// header layout (ints) at start of workspace
#define HDR_ORDER 0      // [128] original batch index per sorted row
#define HDR_DEC   128    // [128] decode length per sorted row (descending)
#define HDR_COUNT 256    // [19]  rows active at step t
#define HDR_OFF   288    // [19]  packed row offset at step t
#define HDR_P     320
#define HDR_T     321
#define HDR_CAP   384    // [128*20] sorted captions
#define HDR_INTS  4096

// ---------------- CDNA5 WMMA types ----------------
typedef __attribute__((ext_vector_type(16))) __bf16 v16bf;
typedef __attribute__((ext_vector_type(8)))  float  v8f;

union ABFrag { v16bf v; uint4 q[2]; };

__device__ __forceinline__ unsigned short f2bf(float f) {
  unsigned u = __float_as_uint(f);
  u += 0x7FFFu + ((u >> 16) & 1u);           // round-to-nearest-even
  return (unsigned short)(u >> 16);
}

// ---------------- prep: sort, packed layout, targets (1 thread, tiny) ------
__global__ void prep_kernel(const int* __restrict__ cap_len,
                            const int* __restrict__ caption,
                            int* __restrict__ hdr,
                            float* __restrict__ target_out) {
  if (threadIdx.x != 0 || blockIdx.x != 0) return;
  // stable counting sort, descending by length (lengths in [2, MAXLEN])
  int idx = 0;
  for (int L = MAXLEN_; L >= 2; --L)
    for (int b = 0; b < B_; ++b)
      if (cap_len[b] == L) { hdr[HDR_ORDER + idx] = b; hdr[HDR_DEC + idx] = L - 1; ++idx; }
  hdr[HDR_T] = hdr[HDR_DEC + 0];
  int off = 0;
  for (int t = 0; t < MAXLEN_ - 1; ++t) {
    int c = 0;
    for (int b = 0; b < B_; ++b) if (hdr[HDR_DEC + b] > t) ++c;
    hdr[HDR_COUNT + t] = c;
    hdr[HDR_OFF + t] = off;
    off += c;
  }
  hdr[HDR_P] = off;
  for (int b = 0; b < B_; ++b) {
    int ob = hdr[HDR_ORDER + b];
    for (int s = 0; s < MAXLEN_; ++s)
      hdr[HDR_CAP + b * MAXLEN_ + s] = caption[ob * MAXLEN_ + s];
  }
  // targets: cap[:,1:][packed order], stored as float
  for (int t = 0; t < MAXLEN_ - 1; ++t) {
    int c = hdr[HDR_COUNT + t], o = hdr[HDR_OFF + t];
    for (int b = 0; b < c; ++b)
      target_out[o + b] = (float)hdr[HDR_CAP + b * MAXLEN_ + t + 1];
  }
}

// ---------------- weight conversion ----------------
__global__ void cvt_bf16_kernel(const float* __restrict__ src,
                                unsigned short* __restrict__ dst, size_t n) {
  size_t i = (size_t)blockIdx.x * blockDim.x + threadIdx.x;
  if (i < n) dst[i] = f2bf(src[i]);
}

// dst[c*rows + r] = src[r*cols + c]   (produce [N,K] from [K,N])
__global__ void cvt_bf16_T_kernel(const float* __restrict__ src,
                                  unsigned short* __restrict__ dst,
                                  int rows, int cols) {
  size_t i = (size_t)blockIdx.x * blockDim.x + threadIdx.x;
  if (i >= (size_t)rows * cols) return;
  int r = (int)(i / cols), c = (int)(i % cols);
  dst[(size_t)c * rows + r] = f2bf(src[i]);
}

// v_sorted = v[order]; also bf16 copy for the pv GEMM
__global__ void permute_v_kernel(const float* __restrict__ v,
                                 const int* __restrict__ hdr,
                                 float* __restrict__ vs,
                                 unsigned short* __restrict__ v_bf) {
  size_t i = (size_t)blockIdx.x * blockDim.x + threadIdx.x;
  const size_t per = (size_t)K_ * VDIM_;
  if (i >= (size_t)B_ * per) return;
  int b = (int)(i / per);
  size_t rest = i % per;
  float val = v[(size_t)hdr[HDR_ORDER + b] * per + rest];
  vs[i] = val;
  v_bf[i] = f2bf(val);
}

__global__ void init_h_kernel(float* __restrict__ h, unsigned short* __restrict__ h_bf) {
  int i = blockIdx.x * blockDim.x + threadIdx.x;
  if (i < B_ * HID_) { h[i] = 0.0f; h_bf[i] = 0; }
}

// ---------------- bf16 WMMA GEMM: C[M,N] = A[M,K] @ Bm[N,K]^T ---------------
// One wave per 16x64 output strip (NT_=4 16x16 WMMA tiles): the A fragment is
// loaded once per k-step and reused for 4 WMMAs (2.5 b128 loads per WMMA).
// K multiple of 32, M and N multiples of 16. A ragged last strip clamps its
// out-of-range tiles to tile 0 (valid loads, stores suppressed) so the inner
// loop stays branch-free with EXEC all-ones.
// mode: 0 = store+bias  1 = relu  2 = relu*wa[n]  3 = packed masked store (fc)
__global__ __launch_bounds__(32) void gemm_bf16_wmma(
    const unsigned short* __restrict__ A, int lda,
    const unsigned short* __restrict__ Bm, int ldb,
    const float* __restrict__ bias,
    float* __restrict__ C, int ldc,
    int K, int N, int mode,
    const float* __restrict__ wa,
    float* __restrict__ dout,
    const int* __restrict__ hdr, int t) {
  const int lane = threadIdx.x;
  const int half = lane >> 4;      // lane group 0-15 / 16-31
  const int l15  = lane & 15;
  const int tile_m  = blockIdx.y * 16;
  const int tile_n0 = blockIdx.x * (16 * NT_);
  const int ntiles  = min(NT_, (N - tile_n0) >> 4);

  const unsigned short* arow = A + (size_t)(tile_m + l15) * lda;   // A row, M = l15
  const unsigned short* brow[NT_];
#pragma unroll
  for (int tn = 0; tn < NT_; ++tn) {
    const int tnc = (tn < ntiles) ? tn : 0;                         // clamp ragged tiles
    brow[tn] = Bm + (size_t)(tile_n0 + tnc * 16 + l15) * ldb;       // W row, N = l15
  }

  v8f acc[NT_];
#pragma unroll
  for (int tn = 0; tn < NT_; ++tn) acc[tn] = (v8f){0.f,0.f,0.f,0.f,0.f,0.f,0.f,0.f};

  for (int k0 = 0; k0 < K; k0 += 32) {
    // A 16x32: lanes 0-15 hold K=0..15, lanes 16-31 hold K=16..31 (matches B)
    ABFrag a;
    a.q[0] = *(const uint4*)(arow + k0 + half * 16);
    a.q[1] = *(const uint4*)(arow + k0 + half * 16 + 8);
    // B 32x16: lanes 0-15 hold K=0..15, lanes 16-31 hold K=16..31 (pairs/VGPR)
    ABFrag b[NT_];
#pragma unroll
    for (int tn = 0; tn < NT_; ++tn) {
      const uint4* bp = (const uint4*)(brow[tn] + k0 + half * 16);
      b[tn].q[0] = bp[0];
      b[tn].q[1] = bp[1];
    }
    if (k0 + 256 < K) {
#pragma unroll
      for (int tn = 0; tn < NT_; ++tn)
        __builtin_prefetch(brow[tn] + k0 + 256, 0, 1);              // global_prefetch_b8
    }
#pragma unroll
    for (int tn = 0; tn < NT_; ++tn)
      acc[tn] = __builtin_amdgcn_wmma_f32_16x16x32_bf16(
          /*neg_a=*/false, a.v, /*neg_b=*/false, b[tn].v,
          /*c_mod=*/(short)0, acc[tn], /*reuse_a=*/false, /*reuse_b=*/false);
  }

  int cnt = 0, off = 0;
  if (mode == 3) { cnt = hdr[HDR_COUNT + t]; off = hdr[HDR_OFF + t]; }
#pragma unroll
  for (int tn = 0; tn < NT_; ++tn) {
    if (tn >= ntiles) break;
    const int n = tile_n0 + tn * 16 + l15;
    const float bn = bias ? bias[n] : 0.0f;
    const float wn = (mode == 2) ? wa[n] : 1.0f;
#pragma unroll
    for (int r = 0; r < 8; ++r) {
      const int m = tile_m + half * 8 + r;   // C/D layout: VGPR r -> M row
      float val = acc[tn][r] + bn;
      if (mode == 1 || mode == 2) val = fmaxf(val, 0.0f);
      if (mode == 2) val *= wn;
      if (mode == 3) {
        if (m < cnt) dout[(size_t)(off + m) * NTOKEN_ + n] = val;
      } else {
        C[(size_t)m * ldc + n] = val;
      }
    }
  }
}

// ---------------- attention + xin build (one block per batch row) ----------
__global__ __launch_bounds__(128) void attn_xin_kernel(
    const float* __restrict__ pv, const float* __restrict__ qw,
    const float* __restrict__ vs, const float* __restrict__ emb,
    const float* __restrict__ ba, const int* __restrict__ hdr,
    unsigned short* __restrict__ xin, int t) {
  __shared__ float qs[HID_];
  __shared__ float e[K_];
  const int b = blockIdx.x;
  const int tid = threadIdx.x;
  for (int h = tid; h < HID_; h += 128) qs[h] = qw[b * HID_ + h];
  if (tid < K_) e[tid] = 0.0f;
  __syncthreads();

  const float* pvb = pv + (size_t)b * K_ * HID_;
  for (int k = 0; k < K_; ++k) {
    const float* row = pvb + k * HID_;
    float p = 0.0f;
    for (int h = tid; h < HID_; h += 128) p += row[h] * qs[h];
    atomicAdd(&e[k], p);                    // ds_add_f32
  }
  __syncthreads();
  if (tid == 0) {                            // softmax over K=36 (tiny)
    float b0 = ba[0], mx = -1e30f;
    for (int k = 0; k < K_; ++k) mx = fmaxf(mx, e[k] + b0);
    float s = 0.0f;
    for (int k = 0; k < K_; ++k) { float ex = __expf(e[k] + b0 - mx); e[k] = ex; s += ex; }
    float inv = 1.0f / s;
    for (int k = 0; k < K_; ++k) e[k] *= inv;
  }
  __syncthreads();

  const float* vb = vs + (size_t)b * K_ * VDIM_;
  unsigned short* xb = xin + (size_t)b * XDIM_;
  for (int c = tid; c < VDIM_; c += 128) {   // att_v -> xin[512:2560]
    float s = 0.0f;
    for (int k = 0; k < K_; ++k) s += e[k] * vb[(size_t)k * VDIM_ + c];
    xb[EMBED_ + c] = f2bf(s);
  }
  const int tok = hdr[HDR_CAP + b * MAXLEN_ + t];
  const float* er = emb + (size_t)tok * EMBED_;
  for (int c = tid; c < EMBED_; c += 128) xb[c] = f2bf(er[c]);  // xt -> xin[0:512]
}

// ---------------- GRU gates + masked state update ----------------
__global__ void gates_kernel(const float* __restrict__ gi, const float* __restrict__ gh,
                             float* __restrict__ h, unsigned short* __restrict__ h_bf,
                             const int* __restrict__ hdr, int t) {
  int idx = blockIdx.x * blockDim.x + threadIdx.x;
  if (idx >= B_ * HID_) return;
  const int b = idx >> 10, hc = idx & (HID_ - 1);
  const float* gib = gi + (size_t)b * G3_;
  const float* ghb = gh + (size_t)b * G3_;
  const float r = 1.0f / (1.0f + __expf(-(gib[hc] + ghb[hc])));
  const float z = 1.0f / (1.0f + __expf(-(gib[HID_ + hc] + ghb[HID_ + hc])));
  const float n = tanhf(gib[2 * HID_ + hc] + r * ghb[2 * HID_ + hc]);
  const float hold = h[idx];
  const float hnew = (1.0f - z) * n + z * hold;
  const float hv = (t < hdr[HDR_DEC + b]) ? hnew : hold;  // freeze finished rows
  h[idx] = hv;
  h_bf[idx] = f2bf(hv);
}

// ---------------- in-place softmax over the TIME dim (faithful to ref) -----
// 20 slots per (b,token): dec_b fc values + (20-dec_b) implicit zeros.
__global__ void time_softmax_kernel(float* __restrict__ pred, const int* __restrict__ hdr) {
  size_t idx = (size_t)blockIdx.x * blockDim.x + threadIdx.x;
  if (idx >= (size_t)B_ * NTOKEN_) return;
  const int b = (int)(idx / NTOKEN_);
  const int tok = (int)(idx % NTOKEN_);
  const int d = hdr[HDR_DEC + b];
  float vals[MAXLEN_ - 1];
  float mx = 0.0f;  // implicit zero slots participate in the max
  for (int t = 0; t < d; ++t) {
    float v = pred[(size_t)(hdr[HDR_OFF + t] + b) * NTOKEN_ + tok];
    vals[t] = v;
    mx = fmaxf(mx, v);
  }
  float denom = (float)(MAXLEN_ - d) * __expf(-mx);
  for (int t = 0; t < d; ++t) denom += __expf(vals[t] - mx);
  const float inv = 1.0f / denom;
  for (int t = 0; t < d; ++t)
    pred[(size_t)(hdr[HDR_OFF + t] + b) * NTOKEN_ + tok] = __expf(vals[t] - mx) * inv;
}

// ---------------- launcher ----------------
extern "C" void kernel_launch(void* const* d_in, const int* in_sizes, int n_in,
                              void* d_out, int out_size, void* d_ws, size_t ws_size,
                              hipStream_t stream) {
  const float* v       = (const float*)d_in[0];
  const int*   caption = (const int*)d_in[1];
  const int*   cap_len = (const int*)d_in[2];
  const float* emb     = (const float*)d_in[3];
  const float* Wv      = (const float*)d_in[4];
  const float* bv      = (const float*)d_in[5];
  const float* Wq      = (const float*)d_in[6];
  const float* bq      = (const float*)d_in[7];
  const float* wa      = (const float*)d_in[8];
  const float* ba      = (const float*)d_in[9];
  const float* Wih     = (const float*)d_in[10];
  const float* Whh     = (const float*)d_in[11];
  const float* bih     = (const float*)d_in[12];
  const float* bhh     = (const float*)d_in[13];
  const float* Wfc     = (const float*)d_in[14];
  const float* bfc     = (const float*)d_in[15];
  float* out = (float*)d_out;

  const int P = out_size / (NTOKEN_ + 1);   // predict P*NTOKEN + target P

  // -------- workspace carve-out (256B aligned) --------
  char* ws = (char*)d_ws;
  size_t o = 0;
  auto carve = [&](size_t bytes) -> void* {
    void* p = ws + o;
    o += (bytes + 255) & ~(size_t)255;
    return p;
  };
  int* hdr = (int*)carve(HDR_INTS * sizeof(int));
  unsigned short* WvT_bf = (unsigned short*)carve((size_t)HID_ * VDIM_ * 2);     // [N=HID,K=VDIM]
  unsigned short* WqT_bf = (unsigned short*)carve((size_t)HID_ * HID_ * 2);      // [N=HID,K=HID]
  unsigned short* Wih_bf = (unsigned short*)carve((size_t)G3_ * XDIM_ * 2);      // already [N,K]
  unsigned short* Whh_bf = (unsigned short*)carve((size_t)G3_ * HID_ * 2);
  unsigned short* Wfc_bf = (unsigned short*)carve((size_t)NTOKEN_ * HID_ * 2);   // 41MB: L2-resident
  unsigned short* v_bf   = (unsigned short*)carve((size_t)B_ * K_ * VDIM_ * 2);
  float* v_sorted        = (float*)carve((size_t)B_ * K_ * VDIM_ * 4);
  float* pv              = (float*)carve((size_t)B_ * K_ * HID_ * 4);
  float* h               = (float*)carve((size_t)B_ * HID_ * 4);
  unsigned short* h_bf   = (unsigned short*)carve((size_t)B_ * HID_ * 2);
  float* qw              = (float*)carve((size_t)B_ * HID_ * 4);
  unsigned short* xin_bf = (unsigned short*)carve((size_t)B_ * XDIM_ * 2);
  float* gi              = (float*)carve((size_t)B_ * G3_ * 4);
  float* gh              = (float*)carve((size_t)B_ * G3_ * 4);
  (void)ws_size; (void)n_in; (void)in_sizes;

  // grid helper for the strip GEMM: x covers N in 64-col strips, y covers M
  auto gemm_grid = [](int M, int N) {
    return dim3((unsigned)((N / 16 + NT_ - 1) / NT_), (unsigned)(M / 16));
  };

  // -------- setup --------
  prep_kernel<<<1, 1, 0, stream>>>(cap_len, caption, hdr, out + (size_t)P * NTOKEN_);
  init_h_kernel<<<(B_ * HID_ + 255) / 256, 256, 0, stream>>>(h, h_bf);

  {
    size_t n;
    n = (size_t)VDIM_ * HID_;
    cvt_bf16_T_kernel<<<(unsigned)((n + 255) / 256), 256, 0, stream>>>(Wv, WvT_bf, VDIM_, HID_);
    n = (size_t)HID_ * HID_;
    cvt_bf16_T_kernel<<<(unsigned)((n + 255) / 256), 256, 0, stream>>>(Wq, WqT_bf, HID_, HID_);
    n = (size_t)G3_ * XDIM_;
    cvt_bf16_kernel<<<(unsigned)((n + 255) / 256), 256, 0, stream>>>(Wih, Wih_bf, n);
    n = (size_t)G3_ * HID_;
    cvt_bf16_kernel<<<(unsigned)((n + 255) / 256), 256, 0, stream>>>(Whh, Whh_bf, n);
    n = (size_t)NTOKEN_ * HID_;
    cvt_bf16_kernel<<<(unsigned)((n + 255) / 256), 256, 0, stream>>>(Wfc, Wfc_bf, n);
  }
  {
    size_t n = (size_t)B_ * K_ * VDIM_;
    permute_v_kernel<<<(unsigned)((n + 255) / 256), 256, 0, stream>>>(v, hdr, v_sorted, v_bf);
  }

  // pv = relu(v_sorted @ Wv + bv): M=4608, N=1024, K=2048
  gemm_bf16_wmma<<<gemm_grid(B_ * K_, HID_), 32, 0, stream>>>(
      v_bf, VDIM_, WvT_bf, VDIM_, bv, pv, HID_, VDIM_, HID_, /*mode=*/1,
      nullptr, nullptr, hdr, 0);

  // -------- 19 decode steps (masks make trailing steps no-ops) --------
  for (int t = 0; t < MAXLEN_ - 1; ++t) {
    // qw = relu(h @ Wq + bq) * wa : M=128, N=1024, K=1024
    gemm_bf16_wmma<<<gemm_grid(B_, HID_), 32, 0, stream>>>(
        h_bf, HID_, WqT_bf, HID_, bq, qw, HID_, HID_, HID_, /*mode=*/2,
        wa, nullptr, hdr, t);

    attn_xin_kernel<<<B_, 128, 0, stream>>>(pv, qw, v_sorted, emb, ba, hdr, xin_bf, t);

    // gi = xin @ Wih^T + bih : M=128, N=3072, K=2560
    gemm_bf16_wmma<<<gemm_grid(B_, G3_), 32, 0, stream>>>(
        xin_bf, XDIM_, Wih_bf, XDIM_, bih, gi, G3_, XDIM_, G3_, /*mode=*/0,
        nullptr, nullptr, hdr, t);
    // gh = h @ Whh^T + bhh : M=128, N=3072, K=1024
    gemm_bf16_wmma<<<gemm_grid(B_, G3_), 32, 0, stream>>>(
        h_bf, HID_, Whh_bf, HID_, bhh, gh, G3_, HID_, G3_, /*mode=*/0,
        nullptr, nullptr, hdr, t);

    gates_kernel<<<(B_ * HID_ + 255) / 256, 256, 0, stream>>>(gi, gh, h, h_bf, hdr, t);

    // fc: out = h_new @ Wfc^T + bfc, written straight into packed predict rows
    gemm_bf16_wmma<<<gemm_grid(B_, NTOKEN_), 32, 0, stream>>>(
        h_bf, HID_, Wfc_bf, HID_, bfc, nullptr, 0, HID_, NTOKEN_, /*mode=*/3,
        nullptr, out, hdr, t);
  }

  // softmax over the time dimension, in place on the packed predict buffer
  {
    size_t n = (size_t)B_ * NTOKEN_;
    time_softmax_kernel<<<(unsigned)((n + 255) / 256), 256, 0, stream>>>(out, hdr);
  }
}
